// QAttention_85693187489862
// MI455X (gfx1250) — matrix operand order repaired and checked by reading, not compiled, not run on hardware
//
#include <hip/hip_runtime.h>
#include <hip/hip_bf16.h>
#include <math.h>

// ---------------------------------------------------------------------------
// Quaternion attention for MI455X (gfx1250, wave32, WMMA).
//  pipeline: pack x -> bf16 | expand Hamilton weights -> dense bf16 |
//            WMMA GEMM qkv (scatter to [3][B][H][N][d]) | flash attention |
//            WMMA GEMM proj | depthwise-PE + transpose epilogue.
// All matmul FLOPs (~103 GFLOP) run on v_wmma_f32_16x16x32_bf16; attention is
// streaming (online softmax) so N*N scores are never materialized in HBM.
// GEMM tiles are staged with GLOBAL_LOAD_ASYNC_TO_LDS_B128 (ASYNCcnt) into
// double-buffered LDS so HBM latency hides behind the WMMA stream.
// ---------------------------------------------------------------------------

typedef __attribute__((ext_vector_type(16))) __bf16 v16bf;
typedef __attribute__((ext_vector_type(8)))  __bf16 v8bf;
typedef __attribute__((ext_vector_type(8)))  float  v8f;
typedef __attribute__((ext_vector_type(4)))  int    v4i;

#if __has_builtin(__builtin_amdgcn_global_load_async_to_lds_b128) && \
    __has_builtin(__builtin_amdgcn_s_wait_asynccnt)
#define HAVE_ASYNC_LDS 1
#else
#define HAVE_ASYNC_LDS 0
#endif

// 16-byte global -> LDS copy. Async path bypasses VGPRs (tracked by ASYNCcnt).
__device__ __forceinline__ void cp16_g2s(const unsigned short* g, unsigned short* s) {
#if HAVE_ASYNC_LDS
  __builtin_amdgcn_global_load_async_to_lds_b128(
      (__attribute__((address_space(1))) v4i*)g,
      (__attribute__((address_space(3))) v4i*)s, 0, 0);
#else
  *(uint4*)s = *(const uint4*)g;
#endif
}
__device__ __forceinline__ void cp_wait() {
#if HAVE_ASYNC_LDS
  __builtin_amdgcn_s_wait_asynccnt(0);
#endif
}

// Hamilton sign table; IDX[q][p] == q ^ p.
__constant__ float c_sgn[16] = { 1.f,-1.f,-1.f,-1.f,
                                 1.f, 1.f, 1.f,-1.f,
                                 1.f,-1.f, 1.f, 1.f,
                                 1.f, 1.f,-1.f, 1.f };

__device__ __forceinline__ unsigned short f2bf(float f) {
  union { float f; unsigned u; } c; c.f = f;
  unsigned u = c.u;
  unsigned r = (u + 0x7FFFu + ((u >> 16) & 1u)) >> 16;   // RNE
  return (unsigned short)r;
}

__device__ __forceinline__ v8f wmma_bf16(v16bf a, v16bf b, v8f c) {
  // (neg_a, A, neg_b, B, c_mod, C, reuse_a, reuse_b)
  return __builtin_amdgcn_wmma_f32_16x16x32_bf16(false, a, false, b,
                                                 (short)0, c, false, false);
}

// A fragment (16 rows x 32 K, bf16), ISA 7.12.2 layout:
// lane<16: K 0-7 in elems0-7, K 16-23 in elems8-15; lane>=16: K 8-15 / 24-31.
__device__ __forceinline__ v16bf load_a_frag(const unsigned short* base,
                                             int row, int ld, int kbase, int lane) {
  int koff = (lane < 16) ? 0 : 8;
  const unsigned short* p = base + row * ld + kbase + koff;
  union { v16bf v; v8bf h[2]; } u;
  u.h[0] = *(const v8bf*)(p);
  u.h[1] = *(const v8bf*)(p + 16);
  return u.v;
}
// B fragment (32 K x 16 N, bf16): lane = N column (mod 16), same K order as A:
// lane<16: K 0-7 / 16-23; lane>=16: K 8-15 / 24-31.
__device__ __forceinline__ v16bf load_b_frag(const unsigned short* base,
                                             int col, int ld, int kbase, int lane) {
  int koff = (lane < 16) ? 0 : 8;
  const unsigned short* p = base + col * ld + kbase + koff;
  union { v16bf v; v8bf h[2]; } u;
  u.h[0] = *(const v8bf*)(p);
  u.h[1] = *(const v8bf*)(p + 16);
  return u.v;
}

// ---------------------------------------------------------------------------
// x (B,Cc,4,H,W) f32 -> Xb[token][ch] bf16, token=b*1024+n, ch=comp*256+cc
__global__ __launch_bounds__(256)
void pack_x_kernel(const float* __restrict__ x, unsigned short* __restrict__ xb) {
  int idx = blockIdx.x * 256 + threadIdx.x;
  int token = idx >> 10, ch = idx & 1023;
  int b = token >> 10, n = token & 1023;
  int comp = ch >> 8, cc = ch & 255;
  xb[idx] = f2bf(x[(((size_t)(b * 256 + cc) * 4 + comp) << 10) + n]);
}

// Expand Hamilton-structured weight (4,Cout,256) f32 into dense bf16
// W[4*Cout][1024]:  W[q*Cout+oc][p*256+cin] = SGN[q][p]*w[(q^p)][oc][cin]
__global__ __launch_bounds__(256)
void expand_w_kernel(const float* __restrict__ w, unsigned short* __restrict__ wx,
                     int coutc) {
  int idx = blockIdx.x * 256 + threadIdx.x;
  int o = idx >> 10, chn = idx & 1023;
  int q = o / coutc, oc = o - q * coutc;
  int p = chn >> 8, cin = chn & 255;
  float v = c_sgn[q * 4 + p] * w[(((size_t)((q ^ p) * coutc + oc)) << 8) + cin];
  wx[idx] = f2bf(v);
}

// ---------------------------------------------------------------------------
// C[M,N] = A[M,K](bf16) @ B[N,K](bf16)^T, f32 accum.
// 128x128 block, BK=32, 8 waves (4x2), each wave 32x64 (2x4 wmma frags).
// Double-buffered LDS; tiles staged with async global->LDS copies.
// mode 0: scatter+bias into qkv [3][8][8][1024][128] bf16
// mode 1: bias + store f32 row-major [M][N]
__global__ __launch_bounds__(256)
void qgemm_kernel(const unsigned short* __restrict__ A,
                  const unsigned short* __restrict__ Bw,
                  const float* __restrict__ bias,
                  int M, int N, int K, int mode,
                  unsigned short* __restrict__ qkv,
                  float* __restrict__ outf) {
  __shared__ __align__(32) unsigned short As[2][128 * 32];
  __shared__ __align__(32) unsigned short Bs[2][128 * 32];
  int tid = threadIdx.x, lane = tid & 31, wid = tid >> 5;
  int wm = wid & 3, wn = wid >> 2;
  int m0 = blockIdx.y * 128, n0 = blockIdx.x * 128;

#define STAGE_TILE(buf, kt)                                                    \
  do {                                                                         \
    for (int i_ = 0; i_ < 4; ++i_) {                                           \
      int e_ = (i_ * 256 + tid) * 8;                                           \
      int r_ = e_ >> 5, c_ = e_ & 31;                                          \
      cp16_g2s(A  + (size_t)(m0 + r_) * K + (kt) * 32 + c_,                    \
               &As[buf][r_ * 32 + c_]);                                        \
      cp16_g2s(Bw + (size_t)(n0 + r_) * K + (kt) * 32 + c_,                    \
               &Bs[buf][r_ * 32 + c_]);                                        \
    }                                                                          \
  } while (0)

  v8f acc[2][4];
  for (int f = 0; f < 2; ++f)
    for (int g = 0; g < 4; ++g) acc[f][g] = 0.0f;

  int nk = K >> 5;
  STAGE_TILE(0, 0);
  cp_wait();
  __syncthreads();

  for (int kt = 0; kt < nk; ++kt) {
    int cur = kt & 1;
    if (kt + 1 < nk) STAGE_TILE(cur ^ 1, kt + 1);   // overlap with compute

    v16bf af[2], bfr[4];
    for (int f = 0; f < 2; ++f)
      af[f] = load_a_frag(As[cur], wm * 32 + f * 16 + (lane & 15), 32, 0, lane);
    for (int g = 0; g < 4; ++g)
      bfr[g] = load_b_frag(Bs[cur], wn * 64 + g * 16 + (lane & 15), 32, 0, lane);
    for (int f = 0; f < 2; ++f)
      for (int g = 0; g < 4; ++g)
        acc[f][g] = wmma_bf16(af[f], bfr[g], acc[f][g]);

    cp_wait();        // next tile landed in LDS
    __syncthreads();  // everyone done with 'cur', next tile published
  }
#undef STAGE_TILE

  int rofs = (lane < 16) ? 0 : 8;
  for (int f = 0; f < 2; ++f) {
    for (int g = 0; g < 4; ++g) {
      int coln = n0 + wn * 64 + g * 16 + (lane & 15);
      float bv = bias[coln];
      for (int r = 0; r < 8; ++r) {
        int rowm = m0 + wm * 32 + f * 16 + r + rofs;
        float v = acc[f][g][r] + bv;
        if (mode == 0) {
          int comp = coln / 768;
          int j = coln - comp * 768;
          int which = j >> 8, cc = j & 255;
          int ch = comp * 256 + cc;            // position inside 1024-dim space
          int head = ch >> 7, dd = ch & 127;
          int b = rowm >> 10, n = rowm & 1023;
          size_t off = ((((size_t)which * 8 + b) * 8 + head) * 1024 + n) * 128 + dd;
          qkv[off] = f2bf(v);
        } else {
          outf[(size_t)rowm * N + coln] = v;
        }
      }
    }
  }
}

// ---------------------------------------------------------------------------
// Flash attention: one block per (b, h, 128-row query block). 8 waves, each
// owns 16 query rows; KV streamed in blocks of 32 with online softmax.
// Next KV block's global loads are issued before compute (register prefetch).
__global__ __launch_bounds__(256)
void qflash_kernel(const unsigned short* __restrict__ qkv,
                   unsigned short* __restrict__ attn) {
  __shared__ __align__(32) unsigned short Qs[128 * 128];  // [n][d]
  __shared__ __align__(32) unsigned short Ks[32 * 128];   // [m][d]
  __shared__ __align__(32) unsigned short Vt[128 * 32];   // [d][m] (transposed)
  __shared__ __align__(32) unsigned short Ps[8][16 * 32]; // per-wave P tile
  const float scale = 0.08838834764831845f;               // 1/sqrt(128)
  int tid = threadIdx.x, lane = tid & 31, w = tid >> 5;
  int qb = blockIdx.x & 7, bh = blockIdx.x >> 3;
  int b = bh >> 3, h = bh & 7;
  size_t qbase = (((size_t)0 * 8 + b) * 8 + h) * 1024 * 128;
  size_t kbase = (((size_t)1 * 8 + b) * 8 + h) * 1024 * 128;
  size_t vbase = (((size_t)2 * 8 + b) * 8 + h) * 1024 * 128;

  for (int i = 0; i < 8; ++i) {                           // Q block -> LDS (async)
    int e = (i * 256 + tid) * 8;
    int r = e >> 7, c = e & 127;
    cp16_g2s(qkv + qbase + (size_t)(qb * 128 + r) * 128 + c, &Qs[r * 128 + c]);
  }

  // Register-prefetch KV block 0 while Q streams in.
  uint4 kreg[2], vreg[2];
  int pr[2], pc[2];
  for (int i = 0; i < 2; ++i) {
    int e = (i * 256 + tid) * 8;
    pr[i] = e >> 7; pc[i] = e & 127;
    kreg[i] = *(const uint4*)(qkv + kbase + (size_t)pr[i] * 128 + pc[i]);
    vreg[i] = *(const uint4*)(qkv + vbase + (size_t)pr[i] * 128 + pc[i]);
  }
  cp_wait();
  __syncthreads();                                        // Q visible

  float mrow[8], lrow[8];
  v8f o[8];
  for (int r = 0; r < 8; ++r) { mrow[r] = -3.0e38f; lrow[r] = 0.0f; }
  for (int dj = 0; dj < 8; ++dj) o[dj] = 0.0f;

  for (int kb = 0; kb < 32; ++kb) {
    // publish staged registers: K row-major, V transposed
    for (int i = 0; i < 2; ++i) {
      *(uint4*)&Ks[pr[i] * 128 + pc[i]] = kreg[i];
      union { uint4 u; unsigned short s[8]; } vv; vv.u = vreg[i];
      for (int j = 0; j < 8; ++j) Vt[(pc[i] + j) * 32 + pr[i]] = vv.s[j];
    }
    __syncthreads();

    if (kb + 1 < 32) {                                    // issue next block early
      for (int i = 0; i < 2; ++i) {
        size_t roff = (size_t)((kb + 1) * 32 + pr[i]) * 128 + pc[i];
        kreg[i] = *(const uint4*)(qkv + kbase + roff);
        vreg[i] = *(const uint4*)(qkv + vbase + roff);
      }
    }

    // S = Q * Kblk^T : two 16x16 frags, contraction over d=128 (4 wmma each)
    v8f s0 = 0.0f, s1 = 0.0f;
    for (int ks = 0; ks < 4; ++ks) {
      v16bf aq  = load_a_frag(Qs, w * 16 + (lane & 15), 128, ks * 32, lane);
      v16bf bk0 = load_b_frag(Ks,  0 + (lane & 15), 128, ks * 32, lane);
      v16bf bk1 = load_b_frag(Ks, 16 + (lane & 15), 128, ks * 32, lane);
      s0 = wmma_bf16(aq, bk0, s0);
      s1 = wmma_bf16(aq, bk1, s1);
    }

    // Online softmax. C-layout: VGPR r holds row r (lanes 0-15) / r+8 (16-31),
    // so row reductions are width-16 shuffles.
    for (int r = 0; r < 8; ++r) {
      float x0 = s0[r] * scale, x1 = s1[r] * scale;
      float mx = fmaxf(x0, x1);
      for (int off = 1; off < 16; off <<= 1) mx = fmaxf(mx, __shfl_xor(mx, off, 16));
      float mnew  = fmaxf(mrow[r], mx);
      float alpha = __expf(mrow[r] - mnew);
      mrow[r] = mnew;
      float p0 = __expf(x0 - mnew), p1 = __expf(x1 - mnew);
      float rs = p0 + p1;
      for (int off = 1; off < 16; off <<= 1) rs += __shfl_xor(rs, off, 16);
      lrow[r] = lrow[r] * alpha + rs;
      for (int dj = 0; dj < 8; ++dj) o[dj][r] *= alpha;
      int row = r + ((lane < 16) ? 0 : 8);
      Ps[w][row * 32 +      (lane & 15)] = f2bf(p0);
      Ps[w][row * 32 + 16 + (lane & 15)] = f2bf(p1);
    }

    // O += P * Vblk  (A = P 16x32, B = Vt columns, 8 d-frags)
    v16bf ap = load_a_frag(&Ps[w][0], (lane & 15), 32, 0, lane);
    for (int dj = 0; dj < 8; ++dj) {
      v16bf bv = load_b_frag(Vt, dj * 16 + (lane & 15), 32, 0, lane);
      o[dj] = wmma_bf16(ap, bv, o[dj]);
    }
    __syncthreads();
  }

  for (int r = 0; r < 8; ++r) {
    float inv = 1.0f / lrow[r];
    int token = b * 1024 + qb * 128 + w * 16 + r + ((lane < 16) ? 0 : 8);
    for (int dj = 0; dj < 8; ++dj) {
      int ch = h * 128 + dj * 16 + (lane & 15);
      attn[(size_t)token * 1024 + ch] = f2bf(o[dj][r] * inv);
    }
  }
}

// ---------------------------------------------------------------------------
// Depthwise quaternion 3x3 PE + final add + transpose to (B,Cc,4,H,W).
__global__ __launch_bounds__(256)
void qpe_final_kernel(const float* __restrict__ proj,   // [8192][1024]
                      const float* __restrict__ x,      // (8,256,4,32,32)
                      const float* __restrict__ w_pe,   // (4,256,1,3,3)
                      const float* __restrict__ b_pe,   // (4,256)
                      float* __restrict__ out) {
  int idx = blockIdx.x * 256 + threadIdx.x;
  int wcol = idx & 31, hrow = (idx >> 5) & 31;
  int comp = (idx >> 10) & 3, cc = (idx >> 12) & 255, b = idx >> 20;
  float acc = b_pe[comp * 256 + cc];
  for (int p = 0; p < 4; ++p) {
    float sg = c_sgn[comp * 4 + p];
    const float* wp = w_pe + (size_t)((comp ^ p) * 256 + cc) * 9;
    const float* xp = x + (((size_t)(b * 256 + cc) * 4 + p) << 10);
    for (int dy = 0; dy < 3; ++dy) {
      int hh = hrow + dy - 1;
      if (hh < 0 || hh > 31) continue;
      for (int dx = 0; dx < 3; ++dx) {
        int ww = wcol + dx - 1;
        if (ww < 0 || ww > 31) continue;
        acc += sg * wp[dy * 3 + dx] * xp[hh * 32 + ww];
      }
    }
  }
  int token = b * 1024 + hrow * 32 + wcol;
  acc += proj[(size_t)token * 1024 + comp * 256 + cc];
  out[idx] = acc;
}

// ---------------------------------------------------------------------------
extern "C" void kernel_launch(void* const* d_in, const int* in_sizes, int n_in,
                              void* d_out, int out_size, void* d_ws, size_t ws_size,
                              hipStream_t stream) {
  const float* x      = (const float*)d_in[0];
  const float* w_qkv  = (const float*)d_in[1];
  const float* b_qkv  = (const float*)d_in[2];
  const float* w_proj = (const float*)d_in[3];
  const float* b_proj = (const float*)d_in[4];
  const float* w_pe   = (const float*)d_in[5];
  const float* b_pe   = (const float*)d_in[6];
  float* out = (float*)d_out;

  // Workspace layout (lifetimes disjoint => aliased; ~75.5 MB total):
  //  [0,16M)      Xb (bf16 8192x1024)   -> later reused as ATT (same size)
  //  [16M,22M)    W1 expanded (3072x1024 bf16)
  //  [22M,24M)    W2 expanded (1024x1024 bf16)
  //  [24M,72M)    QKV [3][8][8][1024][128] bf16 -> later reused as PRJ (f32 32MB)
  char* ws = (char*)d_ws;
  unsigned short* Xb  = (unsigned short*)(ws + 0);
  unsigned short* ATT = (unsigned short*)(ws + 0);
  unsigned short* W1  = (unsigned short*)(ws + 16777216);
  unsigned short* W2  = (unsigned short*)(ws + 23068672);
  unsigned short* QKV = (unsigned short*)(ws + 25165824);
  float*          PRJ = (float*)         (ws + 25165824);

  pack_x_kernel<<<32768, 256, 0, stream>>>(x, Xb);
  expand_w_kernel<<<12288, 256, 0, stream>>>(w_qkv, W1, 768);
  expand_w_kernel<<<4096, 256, 0, stream>>>(w_proj, W2, 256);

  dim3 g1(24, 64);   // N=3072/128, M=8192/128
  qgemm_kernel<<<g1, 256, 0, stream>>>(Xb, W1, b_qkv, 8192, 3072, 1024, 0, QKV, nullptr);

  qflash_kernel<<<512, 256, 0, stream>>>(QKV, ATT);

  dim3 g2(8, 64);    // N=1024/128, M=8192/128
  qgemm_kernel<<<g2, 256, 0, stream>>>(ATT, W2, b_proj, 8192, 1024, 1024, 1, nullptr, PRJ);

  qpe_final_kernel<<<32768, 256, 0, stream>>>(PRJ, x, w_pe, b_pe, out);
}
